// Cross_TransformerBlock_75428215652948
// MI455X (gfx1250) — matrix operand order repaired and checked by reading, not compiled, not run on hardware
//
#include <hip/hip_runtime.h>
#include <hip/hip_bf16.h>

typedef unsigned short u16;
typedef __attribute__((ext_vector_type(16))) __bf16 bf16x16;
typedef __attribute__((ext_vector_type(8)))  float  f32x8;

#define HH 128
#define WWID 128
#define HW 16384
#define C0 192
#define NB 4
#define NHEADS 4
#define CPH 48

struct alignas(16) Chunk { u16 h[8]; };

__device__ __forceinline__ u16 f2bf(float f) {
  unsigned u = __float_as_uint(f);
  u += 0x7fffu + ((u >> 16) & 1u);
  return (u16)(u >> 16);
}
__device__ __forceinline__ float bf2f(u16 h) {
  return __uint_as_float(((unsigned)h) << 16);
}
__device__ __forceinline__ bf16x16 load_frag(const u16* p0, const u16* p1) {
  union { bf16x16 v; Chunk c[2]; } u;
  u.c[0] = *(const Chunk*)p0;
  u.c[1] = *(const Chunk*)p1;
  return u.v;
}

// ---------------- weight conversion (f32 -> bf16, with padding) --------------
__global__ void k_wconv(const float* __restrict__ kvw, const float* __restrict__ qw,
                        const float* __restrict__ pw,  const float* __restrict__ fiw,
                        const float* __restrict__ fow,
                        u16* wkv, u16* wq, u16* wpo, u16* wfi, u16* wfo) {
  int i = blockIdx.x * 256 + threadIdx.x;
  if (i < 73728) { wkv[i] = f2bf(kvw[i]); return; }
  i -= 73728;
  if (i < 36864) { wq[i] = f2bf(qw[i]); return; }
  i -= 36864;
  if (i < 36864) { wpo[i] = f2bf(pw[i]); return; }
  i -= 36864;
  if (i < 196608) { // ffn_in: [1020,192] -> [1024,192] zero-padded rows
    int r = i / 192, c = i % 192;
    wfi[i] = f2bf(r < 1020 ? fiw[r * 192 + c] : 0.f);
    return;
  }
  i -= 196608;
  if (i < 98304) { // ffn_out: [192,510] -> [192,512] zero-padded cols
    int r = i / 512, c = i % 512;
    wfo[i] = f2bf(c < 510 ? fow[r * 510 + c] : 0.f);
  }
}

// ---------------- LayerNorm over channels, emit pixel-major bf16 -------------
__global__ void k_ln(const float* __restrict__ src, const float* __restrict__ w,
                     const float* __restrict__ bias, u16* __restrict__ dst) {
  __shared__ float tile[32 * 193];
  __shared__ float smu[32], srs[32];
  int t = threadIdx.x;
  int b = blockIdx.x / (HW / 32);
  int n0 = (blockIdx.x % (HW / 32)) * 32;
  const float* sb = src + (size_t)b * C0 * HW + n0;
  #pragma unroll
  for (int i = 0; i < 24; ++i) {
    int u = t + i * 256;
    int c = u >> 5, p = u & 31;
    tile[p * 193 + c] = sb[(size_t)c * HW + p];
  }
  __syncthreads();
  if (t < 32) {
    float s = 0.f, ss = 0.f;
    for (int c = 0; c < C0; ++c) { float v = tile[t * 193 + c]; s += v; ss += v * v; }
    float mu = s * (1.f / 192.f);
    float var = ss * (1.f / 192.f) - mu * mu;
    smu[t] = mu;
    srs[t] = rsqrtf(var + 1e-5f);
  }
  __syncthreads();
  u16* db = dst + ((size_t)b * HW + n0) * C0;
  #pragma unroll
  for (int i = 0; i < 24; ++i) {
    int u = t + i * 256;
    int p = u / 192, c = u % 192;
    float v = (tile[p * 193 + c] - smu[p]) * srs[p] * w[c] + bias[c];
    db[(size_t)p * C0 + c] = f2bf(v);
  }
}

// -------- async stage one 16-pixel x K bf16 tile into LDS (ASYNCcnt) ---------
template <int K>
__device__ __forceinline__ void stage_tile_async(const u16* __restrict__ src,
                                                 u16* dst, int t) {
  constexpr int CH  = 2 * K;            // 16B chunks in tile (16 rows * 2K bytes)
  constexpr int LPT = (CH + 255) / 256; // async loads per thread (uniform)
  constexpr int CPR = K / 8;            // chunks per pixel row
  constexpr int Kp  = K + 16;           // padded LDS pitch (u16)
  #pragma unroll
  for (int i = 0; i < LPT; ++i) {
    int u = t + i * 256;
    if (u >= CH) u = CH - 1;            // clamp: redundant idempotent load
    int p = u / CPR, j = u - p * CPR;
    unsigned ldsOff = (unsigned)(size_t)(dst + p * Kp + j * 8);
    unsigned long long ga = (unsigned long long)(size_t)(src + (size_t)p * K + j * 8);
    asm volatile("global_load_async_to_lds_b128 %0, %1, off"
                 :: "v"(ldsOff), "v"(ga) : "memory");
  }
}

// ---------------- WMMA GEMM, 4 pixel tiles / WG, double-buffered async -------
// Out[b][m][n] (bf16, channel-major) = W[M,K] x Bt (pixel-major bf16)
template <int K>
__global__ void k_gemm_cv(const u16* __restrict__ Bt, const u16* __restrict__ W,
                          u16* __restrict__ Out, int M) {
  constexpr int Kp  = K + 16;
  constexpr int LPT = (2 * K + 255) / 256;
  __shared__ __align__(16) u16 lds[2][16 * Kp];
  int t = threadIdx.x, lane = t & 31, wave = t >> 5;
  int b = blockIdx.x / 256;
  int n0 = (blockIdx.x % 256) * 64;     // 4 tiles of 16 pixels
  int m0 = (blockIdx.y * 8 + wave) * 16;
  int row = lane & 15, hi = lane >> 4;
  const u16* wr = W + (size_t)(m0 + row) * K;

  stage_tile_async<K>(Bt + ((size_t)b * HW + n0) * K, &lds[0][0], t);
  #pragma unroll
  for (int it = 0; it < 4; ++it) {
    if (it < 3) {
      stage_tile_async<K>(Bt + ((size_t)b * HW + n0 + (it + 1) * 16) * K,
                          &lds[(it + 1) & 1][0], t);
      asm volatile("s_wait_asynccnt %0" :: "n"(LPT) : "memory");
    } else {
      asm volatile("s_wait_asynccnt 0x0" ::: "memory");
    }
    __syncthreads();
    if (m0 < M) {
      const u16* br = &lds[it & 1][row * Kp + hi * 16];
      f32x8 acc = {0.f, 0.f, 0.f, 0.f, 0.f, 0.f, 0.f, 0.f};
      #pragma unroll
      for (int kb = 0; kb < K; kb += 32) {
        bf16x16 a = load_frag(wr + kb + hi * 16, wr + kb + hi * 16 + 8);
        bf16x16 bb = load_frag(br + kb, br + kb + 8);
        acc = __builtin_amdgcn_wmma_f32_16x16x32_bf16(false, a, false, bb,
                                                      (short)0, acc, false, false);
      }
      u16* ob = Out + ((size_t)b * M + m0 + hi * 8) * HW + n0 + it * 16 + row;
      #pragma unroll
      for (int j = 0; j < 8; ++j) ob[(size_t)j * HW] = f2bf(acc[j]);
    }
    asm volatile("s_wait_dscnt 0x0" ::: "memory");
    __syncthreads();
  }
}

// ---------------- same, with f32 residual epilogue ---------------------------
template <int K>
__global__ void k_gemm_res(const u16* __restrict__ Bt, const u16* __restrict__ W,
                           const float* __restrict__ Res, float* __restrict__ Out,
                           int M) {
  constexpr int Kp  = K + 16;
  constexpr int LPT = (2 * K + 255) / 256;
  __shared__ __align__(16) u16 lds[2][16 * Kp];
  int t = threadIdx.x, lane = t & 31, wave = t >> 5;
  int b = blockIdx.x / 256;
  int n0 = (blockIdx.x % 256) * 64;
  int m0 = (blockIdx.y * 8 + wave) * 16;
  int row = lane & 15, hi = lane >> 4;
  const u16* wr = W + (size_t)(m0 + row) * K;

  stage_tile_async<K>(Bt + ((size_t)b * HW + n0) * K, &lds[0][0], t);
  #pragma unroll
  for (int it = 0; it < 4; ++it) {
    if (it < 3) {
      stage_tile_async<K>(Bt + ((size_t)b * HW + n0 + (it + 1) * 16) * K,
                          &lds[(it + 1) & 1][0], t);
      asm volatile("s_wait_asynccnt %0" :: "n"(LPT) : "memory");
    } else {
      asm volatile("s_wait_asynccnt 0x0" ::: "memory");
    }
    __syncthreads();
    if (m0 < M) {
      const u16* br = &lds[it & 1][row * Kp + hi * 16];
      f32x8 acc = {0.f, 0.f, 0.f, 0.f, 0.f, 0.f, 0.f, 0.f};
      #pragma unroll
      for (int kb = 0; kb < K; kb += 32) {
        bf16x16 a = load_frag(wr + kb + hi * 16, wr + kb + hi * 16 + 8);
        bf16x16 bb = load_frag(br + kb, br + kb + 8);
        acc = __builtin_amdgcn_wmma_f32_16x16x32_bf16(false, a, false, bb,
                                                      (short)0, acc, false, false);
      }
      size_t o = ((size_t)b * M + m0 + hi * 8) * HW + n0 + it * 16 + row;
      const float* rb = Res + o;
      float* ob = Out + o;
      #pragma unroll
      for (int j = 0; j < 8; ++j) ob[(size_t)j * HW] = rb[(size_t)j * HW] + acc[j];
    }
    asm volatile("s_wait_dscnt 0x0" ::: "memory");
    __syncthreads();
  }
}

// ---------------- depthwise 3x3 on kv (k -> channel-major, v -> pixel-major) -
__global__ void k_dw_kv(const u16* __restrict__ in, const float* __restrict__ w,
                        u16* __restrict__ kout, u16* __restrict__ vt) {
  int gid = blockIdx.x * 256 + threadIdx.x;
  int n = gid & (HW - 1);
  int ch = (gid >> 14) % 384;
  int b = gid / (HW * 384);
  int y = n >> 7, x = n & 127;
  const u16* base = in + ((size_t)b * 384 + ch) * HW;
  const float* wc = w + ch * 9;
  float acc = 0.f;
  #pragma unroll
  for (int ky = 0; ky < 3; ++ky) {
    int yy = y + ky - 1;
    if (yy < 0 || yy >= HH) continue;
    #pragma unroll
    for (int kx = 0; kx < 3; ++kx) {
      int xx = x + kx - 1;
      if (xx < 0 || xx >= WWID) continue;
      acc += wc[ky * 3 + kx] * bf2f(base[yy * WWID + xx]);
    }
  }
  if (ch < C0) {
    kout[((size_t)b * C0 + ch) * HW + n] = f2bf(acc);
  } else {
    int cc = ch - C0;
    int hd = cc / CPH, d = cc % CPH;
    vt[(((size_t)(b * NHEADS + hd)) * HW + n) * 64 + d] = f2bf(acc);
  }
}

// ---------------- depthwise 3x3, channel-major bf16 out ----------------------
__global__ void k_dw_gen(const u16* __restrict__ in, const float* __restrict__ w,
                         u16* __restrict__ out, int C) {
  int gid = blockIdx.x * 256 + threadIdx.x;
  int n = gid & (HW - 1);
  int ch = (gid >> 14) % C;
  int b = gid / (HW * C);
  int y = n >> 7, x = n & 127;
  const u16* base = in + ((size_t)b * C + ch) * HW;
  const float* wc = w + ch * 9;
  float acc = 0.f;
  #pragma unroll
  for (int ky = 0; ky < 3; ++ky) {
    int yy = y + ky - 1;
    if (yy < 0 || yy >= HH) continue;
    #pragma unroll
    for (int kx = 0; kx < 3; ++kx) {
      int xx = x + kx - 1;
      if (xx < 0 || xx >= WWID) continue;
      acc += wc[ky * 3 + kx] * bf2f(base[yy * WWID + xx]);
    }
  }
  out[((size_t)b * C + ch) * HW + n] = f2bf(acc);
}

// ---------------- FFN depthwise + exact GELU gate, pixel-major out -----------
__global__ void k_dw_ffn(const u16* __restrict__ in, const float* __restrict__ w,
                         u16* __restrict__ gt) {
  int gid = blockIdx.x * 256 + threadIdx.x;
  int n = gid & (HW - 1);
  int c = (gid >> 14) % 510;
  int b = gid / (HW * 510);
  int y = n >> 7, x = n & 127;
  const u16* b1 = in + ((size_t)b * 1024 + c) * HW;
  const u16* b2 = in + ((size_t)b * 1024 + c + 510) * HW;
  const float* w1 = w + c * 9;
  const float* w2 = w + (c + 510) * 9;
  float f1 = 0.f, f2 = 0.f;
  #pragma unroll
  for (int ky = 0; ky < 3; ++ky) {
    int yy = y + ky - 1;
    if (yy < 0 || yy >= HH) continue;
    #pragma unroll
    for (int kx = 0; kx < 3; ++kx) {
      int xx = x + kx - 1;
      if (xx < 0 || xx >= WWID) continue;
      int o = yy * WWID + xx;
      f1 += w1[ky * 3 + kx] * bf2f(b1[o]);
      f2 += w2[ky * 3 + kx] * bf2f(b2[o]);
    }
  }
  float g = 0.5f * f1 * (1.f + erff(f1 * 0.70710678118654752f)) * f2;
  gt[((size_t)b * HW + n) * 512 + c] = f2bf(g);
}

// ---------------- per-row sum of squares (for L2 norms) ----------------------
__global__ void k_sumsq(const u16* __restrict__ kbf, const u16* __restrict__ qbf,
                        float* __restrict__ kss, float* __restrict__ qss) {
  __shared__ float red[256];
  int row = blockIdx.x;
  const u16* src = (blockIdx.y ? qbf : kbf) + (size_t)row * HW;
  float s = 0.f;
  for (int i = threadIdx.x; i < HW; i += 256) { float v = bf2f(src[i]); s += v * v; }
  red[threadIdx.x] = s;
  __syncthreads();
  for (int off = 128; off; off >>= 1) {
    if (threadIdx.x < off) red[threadIdx.x] += red[threadIdx.x + off];
    __syncthreads();
  }
  if (threadIdx.x == 0) (blockIdx.y ? qss : kss)[row] = red[0];
}

// ---------------- Gram matrix q.k^T via WMMA, split-K ------------------------
__global__ void k_gram(const u16* __restrict__ qbf, const u16* __restrict__ kbf,
                       float* __restrict__ part) {
  int t = threadIdx.x, lane = t & 31, wave = t >> 5;   // 9 waves
  int bh = blockIdx.x, sp = blockIdx.y;
  int b = bh >> 2, h = bh & 3;
  int mt = wave / 3, nt = wave % 3;
  int row = lane & 15, hi = lane >> 4;
  const u16* qr = qbf + (size_t)(b * C0 + h * CPH + mt * 16 + row) * HW;
  const u16* kr = kbf + (size_t)(b * C0 + h * CPH + nt * 16 + row) * HW;
  f32x8 acc = {0.f, 0.f, 0.f, 0.f, 0.f, 0.f, 0.f, 0.f};
  int nbase = sp * 2048;
  for (int kb = 0; kb < 2048; kb += 32) {
    bf16x16 a = load_frag(qr + nbase + kb + hi * 16, qr + nbase + kb + hi * 16 + 8);
    bf16x16 bb = load_frag(kr + nbase + kb + hi * 16, kr + nbase + kb + hi * 16 + 8);
    acc = __builtin_amdgcn_wmma_f32_16x16x32_bf16(false, a, false, bb,
                                                  (short)0, acc, false, false);
  }
  float* pb = part + ((size_t)(sp * 16 + bh)) * 2304;
  #pragma unroll
  for (int j = 0; j < 8; ++j)
    pb[(mt * 16 + j + hi * 8) * 48 + nt * 16 + row] = acc[j];
}

// ---------------- reduce partials, norms, temp; 4x top-k softmax combine -----
__global__ void k_attn(const float* __restrict__ part, const float* __restrict__ qss,
                       const float* __restrict__ kss, const float* __restrict__ temp,
                       const float* __restrict__ mix, u16* __restrict__ Abuf) {
  __shared__ float att[2304];
  int t = threadIdx.x, bh = blockIdx.x;
  int b = bh >> 2, h = bh & 3;
  float tp = temp[h];
  for (int i = t; i < 2304; i += 64) {
    float s = 0.f;
    for (int sp = 0; sp < 8; ++sp) s += part[((size_t)(sp * 16 + bh)) * 2304 + i];
    int c = i / 48, d = i % 48;
    float qn = fmaxf(sqrtf(qss[b * C0 + h * CPH + c]), 1e-12f);
    float kn = fmaxf(sqrtf(kss[b * C0 + h * CPH + d]), 1e-12f);
    att[i] = s * tp / (qn * kn);
  }
  __syncthreads();
  if (t < 48) {
    const float* r = att + t * 48;
    const int kk0 = 24, kk1 = 32, kk2 = 36, kk3 = 38;
    float mx = -1e30f;
    for (int d = 0; d < 48; ++d) mx = fmaxf(mx, r[d]);
    float th0 = mx, th1 = mx, th2 = mx, th3 = mx;
    for (int d = 0; d < 48; ++d) {
      float v = r[d];
      int cg = 0, ce = 0;
      for (int e = 0; e < 48; ++e) { cg += (r[e] > v); ce += (r[e] == v); }
      int hi2 = cg + ce;
      if (cg < kk0 && kk0 <= hi2) th0 = v;
      if (cg < kk1 && kk1 <= hi2) th1 = v;
      if (cg < kk2 && kk2 <= hi2) th2 = v;
      if (cg < kk3 && kk3 <= hi2) th3 = v;
    }
    float S0 = 0.f, S1 = 0.f, S2 = 0.f, S3 = 0.f;
    for (int d = 0; d < 48; ++d) {
      float e = __expf(r[d] - mx);
      if (r[d] >= th0) S0 += e;
      if (r[d] >= th1) S1 += e;
      if (r[d] >= th2) S2 += e;
      if (r[d] >= th3) S3 += e;
    }
    float m0 = mix[0] / S0, m1 = mix[1] / S1, m2 = mix[2] / S2, m3 = mix[3] / S3;
    u16* ab = Abuf + ((size_t)bh * 48 + t) * 64;
    for (int d = 0; d < 48; ++d) {
      float e = __expf(r[d] - mx);
      float wv = 0.f;
      if (r[d] >= th0) wv += m0 * e;
      if (r[d] >= th1) wv += m1 * e;
      if (r[d] >= th2) wv += m2 * e;
      if (r[d] >= th3) wv += m3 * e;
      ab[d] = f2bf(wv);
    }
    for (int d = 48; d < 64; ++d) ab[d] = 0;
  }
}

// ---------------- out = A(48x64) x V(64xHW) via WMMA, pixel-major bf16 out ---
__global__ void k_outv(const u16* __restrict__ Abuf, const u16* __restrict__ vt,
                       u16* __restrict__ outt) {
  int t = threadIdx.x, lane = t & 31, wave = t >> 5;   // 6 waves
  int bh = blockIdx.y, b = bh >> 2, h = bh & 3;
  int mt = wave % 3, ns = wave / 3;
  int n0 = blockIdx.x * 32 + ns * 16;
  int row = lane & 15, hi = lane >> 4;
  const u16* ar = Abuf + ((size_t)bh * 48 + mt * 16 + row) * 64;
  const u16* vr = vt + ((size_t)bh * HW + n0 + row) * 64;
  f32x8 acc = {0.f, 0.f, 0.f, 0.f, 0.f, 0.f, 0.f, 0.f};
  #pragma unroll
  for (int kb = 0; kb < 64; kb += 32) {
    bf16x16 a = load_frag(ar + kb + hi * 16, ar + kb + hi * 16 + 8);
    bf16x16 bb = load_frag(vr + kb + hi * 16, vr + kb + hi * 16 + 8);
    acc = __builtin_amdgcn_wmma_f32_16x16x32_bf16(false, a, false, bb,
                                                  (short)0, acc, false, false);
  }
  u16* ob = outt + ((size_t)b * HW + n0 + row) * C0 + h * CPH + mt * 16 + hi * 8;
  Chunk ch;
  #pragma unroll
  for (int j = 0; j < 8; ++j) ch.h[j] = f2bf(acc[j]);
  *(Chunk*)ob = ch;
}

// ============================================================================
extern "C" void kernel_launch(void* const* d_in, const int* in_sizes, int n_in,
                              void* d_out, int out_size, void* d_ws, size_t ws_size,
                              hipStream_t stream) {
  (void)in_sizes; (void)n_in; (void)out_size; (void)ws_size;
  const float* x     = (const float*)d_in[0];
  const float* y     = (const float*)d_in[1];
  const float* ln1w  = (const float*)d_in[2];
  const float* ln1b  = (const float*)d_in[3];
  const float* ln2w  = (const float*)d_in[4];
  const float* ln2b  = (const float*)d_in[5];
  const float* kvw   = (const float*)d_in[6];
  const float* kvdw  = (const float*)d_in[7];
  const float* qw    = (const float*)d_in[8];
  const float* qdw   = (const float*)d_in[9];
  const float* pw    = (const float*)d_in[10];
  const float* temp  = (const float*)d_in[11];
  const float* mix   = (const float*)d_in[12];
  const float* fiw   = (const float*)d_in[13];
  const float* fdw   = (const float*)d_in[14];
  const float* fow   = (const float*)d_in[15];
  float* out = (float*)d_out;

  char* ws = (char*)d_ws;
  u16*   xn   = (u16*)(ws + 0);            // [NPIX][192] bf16
  u16*   yn   = (u16*)(ws + 25165824);
  u16*   kvp  = (u16*)(ws + 50331648);     // [B][384][HW] bf16
  u16*   qp   = (u16*)(ws + 100663296);    // [B][192][HW] bf16
  u16*   kbf  = (u16*)(ws + 125829120);
  u16*   qbf  = (u16*)(ws + 150994944);
  u16*   vt   = (u16*)(ws + 176160768);    // [16][HW][64] bf16
  u16*   outt = (u16*)(ws + 209715200);    // [NPIX][192] bf16
  float* x2   = (float*)(ws + 234881024);  // [B][192][HW] f32
  char*  sm   = ws + 285212672;
  float* qss  = (float*)(sm);
  float* kss  = (float*)(sm + 3072);
  float* part = (float*)(sm + 6144);
  u16*   Abuf = (u16*)(sm + 1185792);
  u16*   wkv  = (u16*)(sm + 1284096);
  u16*   wq   = (u16*)(sm + 1431552);
  u16*   wpo  = (u16*)(sm + 1505280);
  u16*   wfi  = (u16*)(sm + 1579008);
  u16*   wfo  = (u16*)(sm + 1972224);
  u16*   ffp  = qp;   // alias: FFN pre-activation reuses dead attention bufs
  u16*   gt   = yn;   // alias
  u16*   xn2  = xn;   // alias

  // weights -> bf16 (padded)
  k_wconv<<<1728, 256, 0, stream>>>(kvw, qw, pw, fiw, fow, wkv, wq, wpo, wfi, wfo);
  // LN1(x) -> xn_t ; LN1(y) -> yn_t
  k_ln<<<2048, 256, 0, stream>>>(x, ln1w, ln1b, xn);
  k_ln<<<2048, 256, 0, stream>>>(y, ln1w, ln1b, yn);
  // kv = Wkv @ xn ; qpre = Wq @ yn   (WMMA + async double-buffered LDS stage)
  k_gemm_cv<192><<<dim3(1024, 3), 256, 0, stream>>>(xn, wkv, kvp, 384);
  k_gemm_cv<192><<<dim3(1024, 2), 256, 0, stream>>>(yn, wq, qp, 192);
  // depthwise 3x3
  k_dw_kv<<<98304, 256, 0, stream>>>(kvp, kvdw, kbf, vt);
  k_dw_gen<<<49152, 256, 0, stream>>>(qp, qdw, qbf, 192);
  // L2 norms of q,k rows
  k_sumsq<<<dim3(768, 2), 256, 0, stream>>>(kbf, qbf, kss, qss);
  // Gram q.k^T (split-K WMMA), then 4x top-k softmax combine
  k_gram<<<dim3(16, 8), 288, 0, stream>>>(qbf, kbf, part);
  k_attn<<<16, 64, 0, stream>>>(part, qss, kss, temp, mix, Abuf);
  // out = A @ V (WMMA), pixel-major bf16
  k_outv<<<dim3(512, 16), 192, 0, stream>>>(Abuf, vt, outt);
  // x2 = x + Wpo @ out (WMMA + residual, f32 channel-major)
  k_gemm_res<192><<<dim3(1024, 2), 256, 0, stream>>>(outt, wpo, x, x2, 192);
  // FFN
  k_ln<<<2048, 256, 0, stream>>>(x2, ln2w, ln2b, xn2);
  k_gemm_cv<192><<<dim3(1024, 8), 256, 0, stream>>>(xn2, wfi, ffp, 1024);
  k_dw_ffn<<<130560, 256, 0, stream>>>(ffp, fdw, gt);
  // out = x2 + Wfo @ g (WMMA + residual) -> final [B,192,H,W] f32
  k_gemm_res<512><<<dim3(1024, 2), 256, 0, stream>>>(gt, wfo, x2, out, 192);
}
